// Decoder_82231443849250
// MI455X (gfx1250) — matrix operand order repaired and checked by reading, not compiled, not run on hardware
//
#include <hip/hip_runtime.h>
#include <hip/hip_bf16.h>

// ---------------- problem constants ----------------
constexpr int BSZ = 256, SEQ = 128, H = 1024, J = 12, LAT = 64, OBS = 51;
constexpr int NL = 13, PD = (NL + 1) * 3;      // 42
constexpr int PIN = 2 * J + OBS + LAT;         // 139
constexpr int PINP = 160;                      // padded to 5 k-tiles of 32
constexpr int KT1 = PINP / 32;                 // 5
constexpr int KTH = H / 32;                    // 32
constexpr float FPS = 30.0f, ACT_SCALE = 0.25f, SMIN = 0.05f, SMAX = 0.5f;

// output offsets (flat element index into d_out, return-tuple order)
constexpr size_t O_MU = 0;
constexpr size_t O_JT = (size_t)BSZ * SEQ * PD;
constexpr size_t O_AC = O_JT + (size_t)BSZ * SEQ * J;
constexpr size_t O_LS = O_AC + (size_t)BSZ * SEQ * J;

// workspace offsets (in dwords)
constexpr size_t W_W1P  = 0;                               // 64*5*256
constexpr size_t W_W2P  = W_W1P  + 64 * KT1 * 256;
constexpr size_t W_WA1P = W_W2P  + 64 * KTH * 256;
constexpr size_t W_WA2P = W_WA1P + 64 * KTH * 256;         // 1 ntile
constexpr size_t W_WOVP = W_WA2P + 1 * KTH * 256;          // 3 ntiles
constexpr size_t W_XP   = W_WOVP + 3 * KTH * 256;
constexpr size_t W_H1P  = W_XP   + 16 * KT1 * 256;
constexpr size_t W_HP   = W_H1P  + 16 * KTH * 256;
constexpr size_t W_UP   = W_HP   + 16 * KTH * 256;
constexpr size_t W_Y1   = W_UP   + 16 * KTH * 256;         // f32 [256][16]
constexpr size_t W_Y2   = W_Y1   + (size_t)BSZ * 16;       // f32 [256][48]
constexpr size_t W_PJ   = W_Y2   + (size_t)BSZ * 48;       // f32 [256][12]
constexpr size_t W_PDQ  = W_PJ   + (size_t)BSZ * J;

typedef __attribute__((ext_vector_type(16))) __bf16 v16bf;
typedef __attribute__((ext_vector_type(8)))  float  v8f;

__device__ __forceinline__ unsigned short f2bf(float f) {
  unsigned int u = __float_as_uint(f);
  unsigned int r = u + 0x7FFFu + ((u >> 16) & 1u);   // round-to-nearest-even
  return (unsigned short)(r >> 16);
}

// ---------------- one-time prep ----------------
__global__ void init_state(const float* __restrict__ defp,
                           float* __restrict__ pj, float* __restrict__ pdq) {
  int idx = blockIdx.x * blockDim.x + threadIdx.x;
  if (idx < BSZ * J) { pj[idx] = defp[idx % J]; pdq[idx] = 0.0f; }
}

// Pack W [Nreal,Kreal] row-major (used as act @ W.T) into B-fragment bf16 layout,
// zero-padded to 16-row / 32-col tiles.
__global__ void pack_B(const float* __restrict__ W, int Nreal, int Kreal, int KT,
                       unsigned int* __restrict__ out) {
  int ntile = blockIdx.x, kt = blockIdx.y, lane = threadIdx.x;
  int n = ntile * 16 + (lane & 15);
  int kb = kt * 32 + (lane >> 4) * 8;
  unsigned int d[8];
#pragma unroll
  for (int i = 0; i < 8; ++i) {
    int k0 = kb + (i < 4 ? 0 : 16) + 2 * (i & 3);
    unsigned int lo = (n < Nreal && k0     < Kreal) ? f2bf(W[(size_t)n * Kreal + k0])     : 0u;
    unsigned int hi = (n < Nreal && k0 + 1 < Kreal) ? f2bf(W[(size_t)n * Kreal + k0 + 1]) : 0u;
    d[i] = lo | (hi << 16);
  }
  size_t base = ((size_t)(ntile * KT + kt) * 32 + lane) * 8;
  *reinterpret_cast<uint4*>(out + base)     = make_uint4(d[0], d[1], d[2], d[3]);
  *reinterpret_cast<uint4*>(out + base + 4) = make_uint4(d[4], d[5], d[6], d[7]);
}

// Pack concat([Wo(3,H); Wv(42,H); pad(3,H)]) into 3 n-tiles of B fragments.
__global__ void pack_WoV(const float* __restrict__ Wo, const float* __restrict__ Wv,
                         unsigned int* __restrict__ out) {
  int ntile = blockIdx.x, kt = blockIdx.y, lane = threadIdx.x;
  int n = ntile * 16 + (lane & 15);
  int kb = kt * 32 + (lane >> 4) * 8;
  unsigned int d[8];
#pragma unroll
  for (int i = 0; i < 8; ++i) {
    int k0 = kb + (i < 4 ? 0 : 16) + 2 * (i & 3);
    float v0 = 0.f, v1 = 0.f;
    if (n < 3)       { v0 = Wo[(size_t)n * H + k0];       v1 = Wo[(size_t)n * H + k0 + 1]; }
    else if (n < 45) { v0 = Wv[(size_t)(n - 3) * H + k0]; v1 = Wv[(size_t)(n - 3) * H + k0 + 1]; }
    d[i] = (unsigned int)f2bf(v0) | ((unsigned int)f2bf(v1) << 16);
  }
  size_t base = ((size_t)(ntile * KTH + kt) * 32 + lane) * 8;
  *reinterpret_cast<uint4*>(out + base)     = make_uint4(d[0], d[1], d[2], d[3]);
  *reinterpret_cast<uint4*>(out + base + 4) = make_uint4(d[4], d[5], d[6], d[7]);
}

// ---------------- per-step: build x in A-fragment layout ----------------
__global__ void pack_x(const float* __restrict__ pj, const float* __restrict__ pdq,
                       const float* __restrict__ obs, const float* __restrict__ z,
                       const float* __restrict__ lower, const float* __restrict__ upper,
                       int t, unsigned int* __restrict__ xp) {
  int mtile = blockIdx.x, kt = blockIdx.y, lane = threadIdx.x;
  int row = mtile * 16 + (lane & 15);
  int half = lane >> 4;
  unsigned int d[8];
#pragma unroll
  for (int i = 0; i < 8; ++i) {
    int k0 = kt * 32 + (i < 4 ? 0 : 16) + half * 8 + 2 * (i & 3);
    float v[2];
#pragma unroll
    for (int s = 0; s < 2; ++s) {
      int k = k0 + s;
      float x = 0.0f;
      if (k < J) {
        float jm = 0.5f * (upper[k] + lower[k]);
        float jr = 0.5f * (upper[k] - lower[k]);
        x = (pj[row * J + k] - jm) / jr;
      } else if (k < 2 * J) {
        x = pdq[row * J + (k - J)];
      } else if (k < 2 * J + OBS) {
        x = obs[((size_t)row * SEQ + t) * OBS + (k - 2 * J)];
      } else if (k < PIN) {
        x = z[(size_t)row * LAT + (k - 2 * J - OBS)];
      }
      v[s] = x;
    }
    d[i] = (unsigned int)f2bf(v[0]) | ((unsigned int)f2bf(v[1]) << 16);
  }
  size_t base = ((size_t)(mtile * KT1 + kt) * 32 + lane) * 8;
  *reinterpret_cast<uint4*>(xp + base)     = make_uint4(d[0], d[1], d[2], d[3]);
  *reinterpret_cast<uint4*>(xp + base + 4) = make_uint4(d[4], d[5], d[6], d[7]);
}

// ---------------- fused GEMM + bias + (LN) + ReLU + repack ----------------
// One block = one 16-row M tile x 1024 cols; 8 waves x 8 N-tiles each.
// Register double-buffered over K so ~18 b128 loads stay in flight per burst
// of 8 WMMAs.
__global__ __launch_bounds__(256) void gemm_fused(
    const unsigned int* __restrict__ Apack, const unsigned int* __restrict__ Bpack,
    const float* __restrict__ bias, const float* __restrict__ gamma,
    const float* __restrict__ beta, unsigned int* __restrict__ outPack,
    int KT, int doLN) {
  __shared__ unsigned short sh[16 * H];
  __shared__ float s_sum[16], s_sq[16];

  const int mtile = blockIdx.x;
  const int tid = threadIdx.x;
  const int wave = tid >> 5, lane = tid & 31;
  const int half = lane >> 4, lan16 = lane & 15;

  if (tid < 16) { s_sum[tid] = 0.0f; s_sq[tid] = 0.0f; }
  __syncthreads();

  const v8f zero = {0.f, 0.f, 0.f, 0.f, 0.f, 0.f, 0.f, 0.f};
  v8f acc[8];
#pragma unroll
  for (int nn = 0; nn < 8; ++nn) acc[nn] = zero;

  const unsigned int* aBase = Apack + ((size_t)mtile * KT * 32 + lane) * 8;
  const unsigned int* bBase = Bpack + ((size_t)wave * 8 * KT * 32 + lane) * 8;

  v16bf aCur = *reinterpret_cast<const v16bf*>(aBase);
  v16bf bCur[8], bNxt[8];
#pragma unroll
  for (int nn = 0; nn < 8; ++nn)
    bCur[nn] = *reinterpret_cast<const v16bf*>(bBase + (size_t)nn * KT * 256);

  for (int kt = 0; kt < KT; ++kt) {
    int ktn = (kt + 1 < KT) ? kt + 1 : kt;
    v16bf aNxt = *reinterpret_cast<const v16bf*>(aBase + (size_t)ktn * 256);
#pragma unroll
    for (int nn = 0; nn < 8; ++nn)
      bNxt[nn] = *reinterpret_cast<const v16bf*>(
          bBase + (size_t)nn * KT * 256 + (size_t)ktn * 256);
#pragma unroll
    for (int nn = 0; nn < 8; ++nn)
      acc[nn] = __builtin_amdgcn_wmma_f32_16x16x32_bf16(
          false, aCur, false, bCur[nn], (short)0, acc[nn], false, false);
    aCur = aNxt;
#pragma unroll
    for (int nn = 0; nn < 8; ++nn) bCur[nn] = bNxt[nn];
  }

  // bias
#pragma unroll
  for (int nn = 0; nn < 8; ++nn) {
    float bn = bias[(wave * 8 + nn) * 16 + lan16];
#pragma unroll
    for (int i = 0; i < 8; ++i) acc[nn][i] += bn;
  }

  if (doLN) {
    float ps[8], pq[8];
#pragma unroll
    for (int i = 0; i < 8; ++i) { ps[i] = 0.f; pq[i] = 0.f; }
#pragma unroll
    for (int nn = 0; nn < 8; ++nn)
#pragma unroll
      for (int i = 0; i < 8; ++i) { float v = acc[nn][i]; ps[i] += v; pq[i] += v * v; }
#pragma unroll
    for (int m = 1; m < 16; m <<= 1)
#pragma unroll
      for (int i = 0; i < 8; ++i) {
        ps[i] += __shfl_xor(ps[i], m, 32);
        pq[i] += __shfl_xor(pq[i], m, 32);
      }
    if (lan16 == 0) {
#pragma unroll
      for (int i = 0; i < 8; ++i) {
        atomicAdd(&s_sum[half * 8 + i], ps[i]);
        atomicAdd(&s_sq[half * 8 + i], pq[i]);
      }
    }
  }
  __syncthreads();

  float mean[8], rstd[8];
  if (doLN) {
#pragma unroll
    for (int i = 0; i < 8; ++i) {
      int m = half * 8 + i;
      float mu = s_sum[m] * (1.0f / (float)H);
      float var = s_sq[m] * (1.0f / (float)H) - mu * mu;
      mean[i] = mu;
      rstd[i] = rsqrtf(var + 1e-5f);
    }
  }

#pragma unroll
  for (int nn = 0; nn < 8; ++nn) {
    int n = (wave * 8 + nn) * 16 + lan16;
    float g = doLN ? gamma[n] : 1.0f;
    float be = doLN ? beta[n] : 0.0f;
#pragma unroll
    for (int i = 0; i < 8; ++i) {
      float v = acc[nn][i];
      if (doLN) v = (v - mean[i]) * rstd[i] * g + be;
      v = v > 0.0f ? v : 0.0f;
      sh[(half * 8 + i) * H + n] = f2bf(v);
    }
  }
  __syncthreads();

  // repack LDS tile into A-fragment layout (32 k-tiles, 4 per wave)
#pragma unroll
  for (int r = 0; r < 4; ++r) {
    int kt = wave + r * 8;
    unsigned int d[8];
#pragma unroll
    for (int i = 0; i < 8; ++i) {
      int k = kt * 32 + (i < 4 ? 0 : 16) + half * 8 + 2 * (i & 3);
      unsigned int lo = sh[lan16 * H + k];
      unsigned int hi = sh[lan16 * H + k + 1];
      d[i] = lo | (hi << 16);
    }
    size_t base = ((size_t)(mtile * KTH + kt) * 32 + lane) * 8;
    *reinterpret_cast<uint4*>(outPack + base)     = make_uint4(d[0], d[1], d[2], d[3]);
    *reinterpret_cast<uint4*>(outPack + base + 4) = make_uint4(d[4], d[5], d[6], d[7]);
  }
}

// ---------------- small WMMA head GEMM: [256,1024] x [1024, NT*16] ----------------
// Block = one M tile, NT waves (one N tile each). Raw accum (bias applied later).
__global__ __launch_bounds__(96) void head_mm(
    const unsigned int* __restrict__ Apack, const unsigned int* __restrict__ Bpack,
    float* __restrict__ outPlain, int NT) {
  const int mtile = blockIdx.x;
  const int wave = threadIdx.x >> 5, lane = threadIdx.x & 31;
  const int half = lane >> 4, lan16 = lane & 15;

  const v8f zero = {0.f, 0.f, 0.f, 0.f, 0.f, 0.f, 0.f, 0.f};
  v8f acc = zero;

  const unsigned int* aBase = Apack + ((size_t)mtile * KTH * 32 + lane) * 8;
  const unsigned int* bBase = Bpack + ((size_t)wave * KTH * 32 + lane) * 8;

  v16bf aCur = *reinterpret_cast<const v16bf*>(aBase);
  v16bf bCur = *reinterpret_cast<const v16bf*>(bBase);
  for (int kt = 0; kt < KTH; ++kt) {
    int ktn = (kt + 1 < KTH) ? kt + 1 : kt;
    v16bf aNxt = *reinterpret_cast<const v16bf*>(aBase + (size_t)ktn * 256);
    v16bf bNxt = *reinterpret_cast<const v16bf*>(bBase + (size_t)ktn * 256);
    acc = __builtin_amdgcn_wmma_f32_16x16x32_bf16(false, aCur, false, bCur,
                                                  (short)0, acc, false, false);
    aCur = aNxt; bCur = bNxt;
  }
  int Nout = NT * 16;
#pragma unroll
  for (int i = 0; i < 8; ++i) {
    int m = half * 8 + i;
    int n = wave * 16 + lan16;
    outPlain[(size_t)(mtile * 16 + m) * Nout + n] = acc[i];
  }
}

// ---------------- pointwise finish: activations, state update, outputs ----------------
__global__ __launch_bounds__(64) void finish(
    const float* __restrict__ y1,   // [256][16]: u@Wa2^T raw
    const float* __restrict__ y2,   // [256][48]: h@[Wo;Wv]^T raw
    const float* __restrict__ ba2, const float* __restrict__ bo,
    const float* __restrict__ bv, const float* __restrict__ alpha_logits,
    const float* __restrict__ fk_W, const float* __restrict__ fk_b,
    const float* __restrict__ pos_mean, const float* __restrict__ pos_std,
    const float* __restrict__ lower, const float* __restrict__ upper,
    const float* __restrict__ defp,
    float* __restrict__ pj, float* __restrict__ pdq,
    float* __restrict__ out, int t) {
  int b = blockIdx.x, p = threadIdx.x;
  __shared__ float cj_sh[J], obj_sh[3];

  if (p < J) {
    float a = tanhf(y1[b * 16 + p] + ba2[p]);
    float al = 1.0f / (1.0f + __expf(-alpha_logits[p]));
    float pjv = pj[b * J + p];
    float tgt = a * ACT_SCALE + defp[p];
    float cj = pjv + al * (tgt - pjv);
    cj = fminf(fmaxf(cj, lower[p]), upper[p]);
    cj_sh[p] = cj;
    pj[b * J + p] = cj;                 // detached carry
    pdq[b * J + p] = (cj - pjv) * FPS;
    out[O_JT + ((size_t)b * SEQ + t) * J + p] = cj;
    out[O_AC + ((size_t)b * SEQ + t) * J + p] = a;
  } else if (p < J + 3) {
    int c = p - J;
    obj_sh[c] = y2[b * 48 + c] + bo[c];
  } else if (p < J + 3 + PD) {
    int q = p - (J + 3);
    float s = y2[b * 48 + 3 + q] + bv[q];
    float sig = SMIN + (SMAX - SMIN) / (1.0f + __expf(-s));
    out[O_LS + ((size_t)b * SEQ + t) * PD + q] = __logf(sig);
  }
  __syncthreads();

  if (p < PD) {
    float v;
    if (p < NL * 3) {
      v = fk_b[p];
#pragma unroll
      for (int j = 0; j < J; ++j) v += fk_W[p * J + j] * cj_sh[j];
    } else {
      v = obj_sh[p - NL * 3];
    }
    out[O_MU + ((size_t)b * SEQ + t) * PD + p] = (v - pos_mean[p]) / pos_std[p];
  }
}

// ---------------- driver ----------------
extern "C" void kernel_launch(void* const* d_in, const int* in_sizes, int n_in,
                              void* d_out, int out_size, void* d_ws, size_t ws_size,
                              hipStream_t stream) {
  const float* z        = (const float*)d_in[0];
  const float* obs_seq  = (const float*)d_in[1];
  const float* W1       = (const float*)d_in[2];
  const float* b1       = (const float*)d_in[3];
  const float* g1       = (const float*)d_in[4];
  const float* be1      = (const float*)d_in[5];
  const float* W2       = (const float*)d_in[6];
  const float* b2       = (const float*)d_in[7];
  const float* g2       = (const float*)d_in[8];
  const float* be2      = (const float*)d_in[9];
  const float* Wa1      = (const float*)d_in[10];
  const float* ba1      = (const float*)d_in[11];
  const float* Wa2      = (const float*)d_in[12];
  const float* ba2      = (const float*)d_in[13];
  const float* Wo       = (const float*)d_in[14];
  const float* bo       = (const float*)d_in[15];
  const float* Wv       = (const float*)d_in[16];
  const float* bv       = (const float*)d_in[17];
  const float* alpha_l  = (const float*)d_in[18];
  const float* fk_W     = (const float*)d_in[19];
  const float* fk_b     = (const float*)d_in[20];
  const float* pos_mean = (const float*)d_in[21];
  const float* pos_std  = (const float*)d_in[22];
  const float* jlower   = (const float*)d_in[23];
  const float* jupper   = (const float*)d_in[24];
  const float* defp     = (const float*)d_in[25];

  unsigned int* ws = (unsigned int*)d_ws;
  float* wsf = (float*)d_ws;
  float* out = (float*)d_out;

  float* pj  = wsf + W_PJ;
  float* pdq = wsf + W_PDQ;

  init_state<<<(BSZ * J + 255) / 256, 256, 0, stream>>>(defp, pj, pdq);
  pack_B<<<dim3(H / 16, KT1), 32, 0, stream>>>(W1, H, PIN, KT1, ws + W_W1P);
  pack_B<<<dim3(H / 16, KTH), 32, 0, stream>>>(W2, H, H, KTH, ws + W_W2P);
  pack_B<<<dim3(H / 16, KTH), 32, 0, stream>>>(Wa1, H, H, KTH, ws + W_WA1P);
  pack_B<<<dim3(1, KTH), 32, 0, stream>>>(Wa2, J, H, KTH, ws + W_WA2P);
  pack_WoV<<<dim3(3, KTH), 32, 0, stream>>>(Wo, Wv, ws + W_WOVP);

  for (int t = 0; t < SEQ; ++t) {
    pack_x<<<dim3(BSZ / 16, KT1), 32, 0, stream>>>(pj, pdq, obs_seq, z,
                                                   jlower, jupper, t, ws + W_XP);
    // h1 = relu(LN(x @ W1.T + b1))
    gemm_fused<<<BSZ / 16, 256, 0, stream>>>(ws + W_XP, ws + W_W1P, b1, g1, be1,
                                             ws + W_H1P, KT1, 1);
    // h  = relu(LN(h1 @ W2.T + b2))
    gemm_fused<<<BSZ / 16, 256, 0, stream>>>(ws + W_H1P, ws + W_W2P, b2, g2, be2,
                                             ws + W_HP, KTH, 1);
    // u  = relu(h @ Wa1.T + ba1)
    gemm_fused<<<BSZ / 16, 256, 0, stream>>>(ws + W_HP, ws + W_WA1P, ba1, nullptr,
                                             nullptr, ws + W_UP, KTH, 0);
    // y1 = u @ Wa2.T (raw), y2 = h @ [Wo;Wv].T (raw)
    head_mm<<<BSZ / 16, 32, 0, stream>>>(ws + W_UP, ws + W_WA2P, wsf + W_Y1, 1);
    head_mm<<<BSZ / 16, 96, 0, stream>>>(ws + W_HP, ws + W_WOVP, wsf + W_Y2, 3);
    finish<<<BSZ, 64, 0, stream>>>(wsf + W_Y1, wsf + W_Y2, ba2, bo, bv, alpha_l,
                                   fk_W, fk_b, pos_mean, pos_std, jlower, jupper,
                                   defp, pj, pdq, out, t);
  }
}
